// Head_58050777973550
// MI455X (gfx1250) — matrix operand order repaired and checked by reading, not compiled, not run on hardware
//
#include <hip/hip_runtime.h>
#include <math.h>

typedef __attribute__((ext_vector_type(16))) _Float16 v16h;
typedef __attribute__((ext_vector_type(8)))  _Float16 h8;
typedef __attribute__((ext_vector_type(8)))  float    v8f;
typedef __attribute__((ext_vector_type(4)))  float    f32x4;

#define B_ 8
#define T_ 2048
#define D_ 1024
#define H_ 64

// workspace layout in _Float16 units
#define WT_ELEMS (3 * H_ * D_)          // 196608
#define QK_ELEMS (B_ * T_ * H_)         // 1048576
#define WT_OFF   0
#define Q_OFF    (WT_OFF + WT_ELEMS)
#define K_OFF    (Q_OFF + QK_ELEMS)
#define V_OFF    (K_OFF + QK_ELEMS)

#define WSTRIDE 40   // halves per LDS row (80 B): bank-conflict-free b128 reads

static __device__ __forceinline__ v8f wmma_f16(v16h a, v16h b, v8f c) {
  return __builtin_amdgcn_wmma_f32_16x16x32_f16(false, a, false, b, (short)0, c,
                                                false, false);
}

// ---------------------------------------------------------------------------
// Kernel 0: convert + transpose weights.  wt[mat][n][k] = W_mat[k][n] (f16)
// Row index (mat*64 + n) is exactly j*16+n for fused B-fragment j = mat*4+nt.
// ---------------------------------------------------------------------------
__global__ void convert_w(const float* __restrict__ Wq,
                          const float* __restrict__ Wk,
                          const float* __restrict__ Wv,
                          _Float16* __restrict__ wt) {
  int idx = blockIdx.x * blockDim.x + threadIdx.x;   // 3*64*1024 exact
  int mat = idx >> 16;
  int rem = idx & 65535;
  int n = rem >> 10;
  int kk = rem & 1023;
  const float* W = (mat == 0) ? Wq : ((mat == 1) ? Wk : Wv);
  wt[idx] = (_Float16)W[kk * H_ + n];
}

// ---------------------------------------------------------------------------
// Kernel 1: fused q|k|v = x @ [Wq|Wk|Wv].  One wave -> 16(M) x 192(N).
// Double-buffered LDS pipeline: the block cooperatively stages the 192x32
// W panel of the NEXT k-step (12 KB) while the 12 WMMAs of the CURRENT
// k-step read B fragments from LDS (ds_load_b128, conflict-free 80B rows).
// x is register-pipelined one k-step ahead.  x read from HBM exactly once.
// grid = 256, block = 128 (4 waves).
// ---------------------------------------------------------------------------
__launch_bounds__(128)
__global__ void qkv_gemm(const float* __restrict__ x,
                         const _Float16* __restrict__ wt,
                         _Float16* __restrict__ qo,
                         _Float16* __restrict__ ko,
                         _Float16* __restrict__ vto) {
  __shared__ __align__(16) _Float16 wbuf[2][192 * WSTRIDE];  // 2 x 15 KB

  const int tid  = threadIdx.x;
  const int lane = tid & 31;
  const int wave = tid >> 5;
  const int mtile = blockIdx.x * 4 + wave;     // 0..1023
  const int m0 = mtile * 16;
  const int l15 = lane & 15;
  const int hi  = lane >> 4;

  const float* xrow = x + (size_t)(m0 + l15) * D_;

  v8f c[12];
#pragma unroll
  for (int j = 0; j < 12; ++j)
#pragma unroll
    for (int r = 0; r < 8; ++r) c[j][r] = 0.0f;

  // ---- preload W panel for kc=0 into buffer 0 (6 x 16B chunks per thread) --
#pragma unroll
  for (int i = 0; i < 6; ++i) {
    const int cid = i * 128 + tid;             // 0..767
    const int rr = cid >> 2, qw = cid & 3;     // row 0..191, 16B quad 0..3
    *(h8*)&wbuf[0][rr * WSTRIDE + qw * 8] =
        *(const h8*)(wt + (size_t)rr * D_ + qw * 8);
  }

  // ---- preload x chunk for kc=0 into registers ----
  f32x4 xr0 = *(const f32x4*)(xrow + hi * 8);
  f32x4 xr1 = *(const f32x4*)(xrow + hi * 8 + 4);
  f32x4 xr2 = *(const f32x4*)(xrow + 16 + hi * 8);
  f32x4 xr3 = *(const f32x4*)(xrow + 16 + hi * 8 + 4);

  __syncthreads();

  for (int kc = 0; kc < D_; kc += 32) {
    const int buf  = (kc >> 5) & 1;
    const int nbuf = buf ^ 1;
    const bool more = (kc + 32) < D_;

    // ---- issue next-chunk global loads early (hide latency under WMMAs) ----
    h8 wstage[6];
    f32x4 xn0, xn1, xn2, xn3;
    if (more) {
#pragma unroll
      for (int i = 0; i < 6; ++i) {
        const int cid = i * 128 + tid;
        const int rr = cid >> 2, qw = cid & 3;
        wstage[i] = *(const h8*)(wt + (size_t)rr * D_ + (kc + 32) + qw * 8);
      }
      xn0 = *(const f32x4*)(xrow + kc + 32 + hi * 8);
      xn1 = *(const f32x4*)(xrow + kc + 32 + hi * 8 + 4);
      xn2 = *(const f32x4*)(xrow + kc + 48 + hi * 8);
      xn3 = *(const f32x4*)(xrow + kc + 48 + hi * 8 + 4);
    }

    // ---- A fragment from pipelined x registers (f32 -> f16) ----
    v16h a;
#pragma unroll
    for (int e = 0; e < 4; ++e) {
      a[e]      = (_Float16)xr0[e];
      a[4 + e]  = (_Float16)xr1[e];
      a[8 + e]  = (_Float16)xr2[e];
      a[12 + e] = (_Float16)xr3[e];
    }

    // ---- 12 WMMAs, B fragments from current LDS buffer ----
#pragma unroll
    for (int j = 0; j < 12; ++j) {
      const _Float16* pb = &wbuf[buf][(j * 16 + l15) * WSTRIDE + hi * 8];
      h8 b0 = *(const h8*)pb;
      h8 b1 = *(const h8*)(pb + 16);
      v16h b;
#pragma unroll
      for (int e = 0; e < 8; ++e) { b[e] = b0[e]; b[8 + e] = b1[e]; }
      c[j] = wmma_f16(a, b, c[j]);
    }

    // ---- commit staged panel to the other buffer, rotate x pipeline ----
    if (more) {
#pragma unroll
      for (int i = 0; i < 6; ++i) {
        const int cid = i * 128 + tid;
        const int rr = cid >> 2, qw = cid & 3;
        *(h8*)&wbuf[nbuf][rr * WSTRIDE + qw * 8] = wstage[i];
      }
      xr0 = xn0; xr1 = xn1; xr2 = xn2; xr3 = xn3;
    }
    __syncthreads();
  }

  // q (j=0..3) and k (j=4..7): [b*T][64] row-major f16
#pragma unroll
  for (int m = 0; m < 2; ++m) {
    _Float16* outm = (m == 0) ? qo : ko;
#pragma unroll
    for (int nt = 0; nt < 4; ++nt)
#pragma unroll
      for (int r = 0; r < 8; ++r)
        outm[(size_t)(m0 + r + 8 * hi) * H_ + nt * 16 + l15] =
            (_Float16)c[m * 4 + nt][r];
  }
  // v (j=8..11): transposed vT[b][h][t], 8 consecutive t per lane -> 16B store
  {
    const int bidx = m0 / T_;
    const int tb   = (m0 % T_) + 8 * hi;
#pragma unroll
    for (int nt = 0; nt < 4; ++nt) {
      const int h = nt * 16 + l15;
      h8 pk;
#pragma unroll
      for (int r = 0; r < 8; ++r) pk[r] = (_Float16)c[8 + nt][r];
      *(h8*)(vto + ((size_t)bidx * H_ + h) * T_ + tb) = pk;
    }
  }
}

// ---------------------------------------------------------------------------
// Kernel 2: flash attention, key-split across the 4 waves of a block.
// Block = one (b, 16-query) tile; wave w takes key-chunks w, w+4, w+8, ...
// Each wave keeps private online-softmax state; partials merged via LDS.
// grid = (128, 8), block = 128.
// ---------------------------------------------------------------------------
__launch_bounds__(128)
__global__ void attn(const _Float16* __restrict__ q,
                     const _Float16* __restrict__ k,
                     const _Float16* __restrict__ vt,
                     float* __restrict__ out) {
  __shared__ __align__(32) float    lO[4][16][64];   // 16 KB partial outputs
  __shared__ float                  lM[4][16];
  __shared__ float                  lL[4][16];
  __shared__ __align__(32) _Float16 lP[4][16][32];   // per-wave P transpose

  const int lane = threadIdx.x & 31;
  const int wave = threadIdx.x >> 5;
  const int b = blockIdx.y;
  const int qtile = blockIdx.x;              // 0..127
  const int q0 = qtile * 16;
  const int l15 = lane & 15;
  const int hi  = lane >> 4;

  const _Float16* qb = q  + (size_t)b * T_ * H_;
  const _Float16* kb = k  + (size_t)b * T_ * H_;
  const _Float16* vb = vt + (size_t)b * H_ * T_;

  // Q A-fragments for h-chunks 0-31 and 32-63
  v16h qa[2];
#pragma unroll
  for (int hc = 0; hc < 2; ++hc) {
    const _Float16* p = qb + (size_t)(q0 + l15) * H_ + hc * 32;
    h8 lo  = *(const h8*)(p + hi * 16);
    h8 hi8 = *(const h8*)(p + hi * 16 + 8);
#pragma unroll
    for (int e = 0; e < 8; ++e) { qa[hc][e] = lo[e]; qa[hc][8 + e] = hi8[e]; }
  }

  v8f o[4];
  float mrow[8], lrow[8];
#pragma unroll
  for (int r = 0; r < 8; ++r) { mrow[r] = -INFINITY; lrow[r] = 0.0f; }
#pragma unroll
  for (int t = 0; t < 4; ++t)
#pragma unroll
    for (int r = 0; r < 8; ++r) o[t][r] = 0.0f;

  const float scale = 45.25483399593904f;    // sqrt(2048): reference quirk
  const int nchunks = (q0 + 16 + 31) / 32;   // 32-key chunks up to diagonal

  for (int ch = wave; ch < nchunks; ch += 4) {
    const int sc = ch * 32;
    __builtin_prefetch(kb + (size_t)(sc + 128 + l15) * H_, 0, 0);

    // ---- stage all 4 K B-fragments, then 4 S-WMMAs ----
    v16h bk[4];
#pragma unroll
    for (int hc = 0; hc < 2; ++hc) {
      bk[hc * 2 + 0] =
          *(const v16h*)(kb + (size_t)(sc + l15) * H_ + hc * 32 + hi * 16);
      bk[hc * 2 + 1] =
          *(const v16h*)(kb + (size_t)(sc + 16 + l15) * H_ + hc * 32 + hi * 16);
    }
    v8f s0, s1;
#pragma unroll
    for (int r = 0; r < 8; ++r) { s0[r] = 0.0f; s1[r] = 0.0f; }
    s0 = wmma_f16(qa[0], bk[0], s0);
    s1 = wmma_f16(qa[0], bk[1], s1);
    s0 = wmma_f16(qa[1], bk[2], s0);
    s1 = wmma_f16(qa[1], bk[3], s1);

    // ---- issue V B-fragment loads early: latency hides under softmax ----
    v16h bv[4];
#pragma unroll
    for (int nt = 0; nt < 4; ++nt)
      bv[nt] =
          *(const v16h*)(vb + (size_t)(nt * 16 + l15) * T_ + sc + hi * 16);

    // ---- scale + causal mask + online softmax (per 8 rows of half-wave) ----
    float p0[8], p1[8];
#pragma unroll
    for (int r = 0; r < 8; ++r) {
      const int qi = q0 + r + 8 * hi;
      float a0 = (sc + l15 <= qi)      ? s0[r] * scale : -INFINITY;
      float a1 = (sc + 16 + l15 <= qi) ? s1[r] * scale : -INFINITY;
      float mx = fmaxf(a0, a1);
#pragma unroll
      for (int off = 1; off < 16; off <<= 1)
        mx = fmaxf(mx, __shfl_xor(mx, off, 32));   // stays within half-wave
      float mnew  = fmaxf(mrow[r], mx);
      float alpha = __expf(mrow[r] - mnew);
      float e0 = __expf(a0 - mnew);
      float e1 = __expf(a1 - mnew);
      float srow = e0 + e1;
#pragma unroll
      for (int off = 1; off < 16; off <<= 1)
        srow += __shfl_xor(srow, off, 32);
      lrow[r] = lrow[r] * alpha + srow;
      mrow[r] = mnew;
      p0[r] = e0; p1[r] = e1;
#pragma unroll
      for (int nt = 0; nt < 4; ++nt) o[nt][r] *= alpha;
    }

    // ---- transpose P (C-layout -> A-layout) through per-wave LDS tile ----
#pragma unroll
    for (int r = 0; r < 8; ++r) {
      lP[wave][r + 8 * hi][l15]      = (_Float16)p0[r];
      lP[wave][r + 8 * hi][16 + l15] = (_Float16)p1[r];
    }
    asm volatile("s_wait_dscnt 0" ::: "memory");   // cross-lane LDS hazard
    v16h pa;
    {
      h8 lo  = *(const h8*)&lP[wave][l15][hi * 16];
      h8 hi8 = *(const h8*)&lP[wave][l15][hi * 16 + 8];
#pragma unroll
      for (int e = 0; e < 8; ++e) { pa[e] = lo[e]; pa[8 + e] = hi8[e]; }
    }

    // ---- O += P @ V ----
#pragma unroll
    for (int nt = 0; nt < 4; ++nt) o[nt] = wmma_f16(pa, bv[nt], o[nt]);
  }

  // ---- publish per-wave partials (unnormalized O, row max m, row sum l) ----
#pragma unroll
  for (int nt = 0; nt < 4; ++nt)
#pragma unroll
    for (int r = 0; r < 8; ++r)
      lO[wave][r + 8 * hi][nt * 16 + l15] = o[nt][r];
  if (l15 == 0) {
#pragma unroll
    for (int r = 0; r < 8; ++r) {
      lM[wave][r + 8 * hi] = mrow[r];
      lL[wave][r + 8 * hi] = lrow[r];
    }
  }
  __syncthreads();

  // ---- merge the 4 key-split partials; 128 threads cover 16x64 outputs ----
  {
    const int row = threadIdx.x >> 3;          // 0..15
    const int cb  = (threadIdx.x & 7) * 8;     // column base, 8 per thread
    float M = lM[0][row];
#pragma unroll
    for (int w = 1; w < 4; ++w) M = fmaxf(M, lM[w][row]);
    float wgt[4];
    float L = 0.0f;
#pragma unroll
    for (int w = 0; w < 4; ++w) {
      wgt[w] = __expf(lM[w][row] - M);         // exp(-inf - M) = 0 for idle waves
      L += wgt[w] * lL[w][row];
    }
    const float inv = 1.0f / L;
    float* ob = out + ((size_t)b * T_ + q0 + row) * H_ + cb;
#pragma unroll
    for (int cc = 0; cc < 8; ++cc) {
      float acc = 0.0f;
#pragma unroll
      for (int w = 0; w < 4; ++w) acc += wgt[w] * lO[w][row][cb + cc];
      ob[cc] = acc * inv;
    }
  }
}

// ---------------------------------------------------------------------------
extern "C" void kernel_launch(void* const* d_in, const int* in_sizes, int n_in,
                              void* d_out, int out_size, void* d_ws,
                              size_t ws_size, hipStream_t stream) {
  const float* x  = (const float*)d_in[0];
  const float* Wq = (const float*)d_in[1];
  const float* Wk = (const float*)d_in[2];
  const float* Wv = (const float*)d_in[3];
  float* out = (float*)d_out;

  _Float16* ws = (_Float16*)d_ws;
  _Float16* wt = ws + WT_OFF;
  _Float16* qf = ws + Q_OFF;
  _Float16* kf = ws + K_OFF;
  _Float16* vf = ws + V_OFF;

  convert_w<<<(3 * H_ * D_) / 256, 256, 0, stream>>>(Wq, Wk, Wv, wt);

  qkv_gemm<<<256, 128, 0, stream>>>(x, wt, qf, kf, vf);

  dim3 g2(128, 8);
  attn<<<g2, 128, 0, stream>>>(qf, kf, vf, out);
}
